// HeterogenousMHA_6442450944106
// MI455X (gfx1250) — matrix operand order repaired and checked by reading, not compiled, not run on hardware
//
#include <hip/hip_runtime.h>

// ---------------------------------------------------------------------------
// Heterogeneous MHA for MI455X (gfx1250, wave32, WMMA).
//   H=8 B=16 G=501 D=128 KD=16 E=128 np=250
// 4-kernel pipeline, one wave32 per workgroup.
// Workspace layout (f16 halves), 7 regions of S = H*B*512*16 = 1M halves,
// plus 16K halves for transposed W_out:
//   [0S) Qmain  [1S) Qap(sel W2/W6)  [2S) Qad(sel W3/W5)  [3S) Qdiag(sel W1/W4)
//   [4S) K (row-major [key][kd])     [5S) V^T ([kd][key]) [6S) heads [b][n][h*16]
//   [7S) WoutT [e][h*16+kd]  (128x128 f16)
// Total 7S+16384 halves ~= 14.03 MB of d_ws.
// ---------------------------------------------------------------------------

#define HH   8
#define BB   16
#define GG   501
#define DD   128
#define KDIM 16
#define EE   128
#define RPAD 512

typedef _Float16 v16h __attribute__((ext_vector_type(16)));
typedef _Float16 v8h  __attribute__((ext_vector_type(8)));
typedef float    v8f  __attribute__((ext_vector_type(8)));

union H16 { v16h v; v8h h[2]; _Float16 e[16]; };
union F8  { v8f  v; float e[8]; };

static __device__ __forceinline__ v8f wmma_f16(v16h a, v16h b, v8f c) {
  // D = A(16x32,f16) * B(32x16,f16) + C(16x16,f32)
  return __builtin_amdgcn_wmma_f32_16x16x32_f16(false, a, false, b, (short)0, c,
                                                false, false);
}

struct WPtrs { const float* w[9]; };

static constexpr size_t SREG = (size_t)HH * BB * RPAD * 16;

// ---------------------------------------------------------------------------
// Kernel 0: transpose W_out (f32 [h*16+kd][e]) -> f16 WoutT [e][h*16+kd].
// ---------------------------------------------------------------------------
__global__ void __launch_bounds__(32) prep_wout(const float* __restrict__ Wout,
                                                _Float16* __restrict__ ws) {
  _Float16* WoutT = ws + 7 * SREG;
  int base = blockIdx.x * 1024;
  for (int idx = base + threadIdx.x; idx < base + 1024; idx += 32) {
    int e = idx >> 7, k = idx & 127;
    WoutT[idx] = (_Float16)Wout[(size_t)k * EE + e];
  }
}

// ---------------------------------------------------------------------------
// Kernel 1: projections. grid = H*B*32 row-tiles, 32 threads.
// For each 16-row tile: 9 GEMMs (16x16x128) = 36 WMMAs, store f16 with
// per-row region selection (pick rows 1..250, delivery rows 251..500).
// ---------------------------------------------------------------------------
__global__ void __launch_bounds__(32) proj_kernel(const float* __restrict__ q,
                                                  WPtrs wp,
                                                  _Float16* __restrict__ ws) {
  __shared__ _Float16 At[16 * 128];   // q rows, f16 (row-major [row][d])
  __shared__ _Float16 Wt[16 * 128];   // current weight TRANSPOSED [kd][d]

  const int l = threadIdx.x;
  const int m = l & 15;
  const int g = l >> 4;
  const int t  = blockIdx.x & 31;
  const int hb = blockIdx.x >> 5;
  const int b  = hb & (BB - 1);
  const int h  = hb >> 4;
  const int r0 = t * 16;

  // stage 16 q rows (f32 -> f16), zero-pad rows >= 501
  for (int idx = l; idx < 16 * 128; idx += 32) {
    int row = idx >> 7, d = idx & 127;
    int grow = r0 + row;
    float v = (grow < GG) ? q[(size_t)(b * GG + grow) * DD + d] : 0.f;
    At[idx] = (_Float16)v;
  }
  __syncthreads();

  // A fragments for the 4 K-chunks (ISA A-layout: e<8 -> K=8g+e, e>=8 -> K=16+8g+e-8)
  H16 afr[4];
#pragma unroll
  for (int c = 0; c < 4; ++c) {
    const _Float16* rp = &At[m * 128 + 32 * c + 8 * g];
    afr[c].h[0] = *(const v8h*)rp;
    afr[c].h[1] = *(const v8h*)(rp + 16);
  }

  const size_t hb512 = (size_t)(h * BB + b) * RPAD;

  for (int w = 0; w < 9; ++w) {
    __syncthreads();
    const float* W = wp.w[w] + (size_t)h * DD * KDIM;
    // stage transposed: Wt[n][d] = W[d][n]
    for (int idx = l; idx < 16 * 128; idx += 32) {
      int n = idx >> 7, d = idx & 127;
      Wt[idx] = (_Float16)W[(size_t)d * KDIM + n];
    }
    __syncthreads();

    F8 acc;
#pragma unroll
    for (int i = 0; i < 8; ++i) acc.e[i] = 0.f;

#pragma unroll
    for (int c = 0; c < 4; ++c) {
      H16 bf;  // B-layout: lane col n=m, e<8 -> K = 32c + 8g + e, e>=8 -> K = 32c + 16 + 8g + e-8
      bf.h[0] = *(const v8h*)(&Wt[m * 128 + 32 * c + 8 * g]);
      bf.h[1] = *(const v8h*)(&Wt[m * 128 + 32 * c + 16 + 8 * g]);
      acc.v = wmma_f16(afr[c].v, bf.v, acc.v);
    }

    // slot / predicate selection (uniform per w)
    size_t off = 0; int needPick = 0, needDel = 0; int isVt = 0;
    switch (w) {
      case 0: off = 0 * SREG; break;                 // W_query -> Qmain
      case 1: off = 4 * SREG; break;                 // W_key   -> K
      case 2: off = 5 * SREG; isVt = 1; break;       // W_val   -> V^T
      case 3: off = 3 * SREG; needPick = 1; break;   // W1 -> diag (pick)
      case 4: off = 1 * SREG; needPick = 1; break;   // W2 -> ap   (pick)
      case 5: off = 2 * SREG; needPick = 1; break;   // W3 -> ad   (pick)
      case 6: off = 3 * SREG; needDel = 1; break;    // W4 -> diag (del)
      case 7: off = 2 * SREG; needDel = 1; break;    // W5 -> ad   (del)
      case 8: off = 1 * SREG; needDel = 1; break;    // W6 -> ap   (del)
    }

#pragma unroll
    for (int j = 0; j < 8; ++j) {
      int grow = r0 + j + 8 * g;                     // < 512 always
      bool pick = (grow >= 1) && (grow <= 250);
      bool del  = (grow >= 251) && (grow <= 500);
      _Float16 hv = (_Float16)((grow < GG) ? acc.e[j] : 0.f);
      bool pred = needPick ? pick : (needDel ? del : true);
      if (isVt) {
        ws[off + ((size_t)(h * BB + b) * 16 + m) * RPAD + grow] = hv;
      } else if (pred) {
        ws[off + (hb512 + grow) * 16 + m] = hv;
      }
    }
  }
}

// ---------------------------------------------------------------------------
// Kernel 2: fused attention with online softmax. grid = H*B*32 query tiles.
// 33 superblocks of 32 keys (all 32-aligned bases, masked to the valid key
// range per block type): 2 score WMMAs + 1 P*V WMMA each, plus one scalar
// "diag" column per row. Row stats live in registers via the D-fragment row
// mapping (row = j + 8*(lane>=16)); 16-lane shfl_xor butterflies reduce
// across columns.
// ---------------------------------------------------------------------------
__global__ void __launch_bounds__(32) attn_kernel(_Float16* __restrict__ ws) {
  __shared__ _Float16 Pt[16 * 32];    // P tile, D-layout -> A-layout bounce

  const int l = threadIdx.x;
  const int m = l & 15;
  const int g = l >> 4;
  const int t  = blockIdx.x & 31;
  const int hb = blockIdx.x >> 5;
  const int b  = hb & (BB - 1);
  const int h  = hb >> 4;
  const int r0 = t * 16;

  const _Float16* Qs  = ws + 0 * SREG;
  const _Float16* Aps = ws + 1 * SREG;
  const _Float16* Ads = ws + 2 * SREG;
  const _Float16* Dgs = ws + 3 * SREG;
  const _Float16* Ks  = ws + 4 * SREG;
  const _Float16* Vts = ws + 5 * SREG;
  _Float16*       Hd  = ws + 6 * SREG;

  const size_t hb512 = (size_t)(h * BB + b) * RPAD;
  const size_t vtb   = ((size_t)(h * BB + b) * 16 + m) * RPAD;
  const int qrow = r0 + m;            // A-fragment row for this lane (<512)

  // Q fragments (A-layout). KD=16 -> only K-halves 0..15 real, e>=8 zero-pad.
  H16 fq, fap, fad;
  {
    fq.h[0]  = *(const v8h*)(Qs  + (hb512 + qrow) * 16 + 8 * g);
    fap.h[0] = *(const v8h*)(Aps + (hb512 + qrow) * 16 + 8 * g);
    fad.h[0] = *(const v8h*)(Ads + (hb512 + qrow) * 16 + 8 * g);
#pragma unroll
    for (int e = 8; e < 16; ++e) { fq.e[e] = (_Float16)0.f; fap.e[e] = (_Float16)0.f; fad.e[e] = (_Float16)0.f; }
  }

  float mrow[8], srow[8];
  F8 acc;
#pragma unroll
  for (int j = 0; j < 8; ++j) { mrow[j] = -1e30f; srow[j] = 0.f; acc.e[j] = 0.f; }

  H16 zh;
#pragma unroll
  for (int e = 0; e < 16; ++e) zh.e[e] = (_Float16)0.f;
  F8 zf;
#pragma unroll
  for (int i = 0; i < 8; ++i) zf.e[i] = 0.f;

  for (int it = 0; it < 33; ++it) {
    int kbase, kstart, kend, qt;             // qt: 0=main 1=ap 2=ad (uniform)
    if (it < 16)      { kbase = 32 * it;              kstart = 0;   kend = GG;  qt = 0; }
    else if (it < 24) { kbase = 32 * (it - 16);       kstart = 1;   kend = 251; qt = 1; }
    else              { kbase = 224 + 32 * (it - 24); kstart = 251; kend = GG;  qt = 2; }

    const H16& fA = (qt == 0) ? fq : ((qt == 1) ? fap : fad);

    // prefetch next superblock's K rows (gfx1250 global_prefetch_b8 path)
    __builtin_prefetch((const void*)(Ks + (hb512 + ((kbase + 32 + m) & (RPAD - 1))) * 16), 0, 1);

    // B fragments: lane col = key, element e<8 -> contraction K(kd)=8g+e;
    // elements e>=8 are the zero K-half (KD=16 padded to 32).
    int key0 = kbase + m;
    int key1 = key0 + 16;                    // both < 512 by construction
    H16 b0, b1;
    b0.h[0] = *(const v8h*)(Ks + (hb512 + key0) * 16 + 8 * g); b0.h[1] = zh.h[0];
    b1.h[0] = *(const v8h*)(Ks + (hb512 + key1) * 16 + 8 * g); b1.h[1] = zh.h[0];

    F8 s0, s1;
    s0.v = wmma_f16(fA.v, b0.v, zf.v);
    s1.v = wmma_f16(fA.v, b1.v, zf.v);

    bool c0 = (key0 >= kstart) && (key0 < kend);
    bool c1 = (key1 >= kstart) && (key1 < kend);

#pragma unroll
    for (int j = 0; j < 8; ++j) {
      int grow = r0 + j + 8 * g;
      bool ra = (qt == 0) ? (grow < GG) : ((grow >= 1) && (grow < GG));
      float x0 = (ra && c0) ? s0.e[j] * 0.25f : -1e30f;
      float x1 = (ra && c1) ? s1.e[j] * 0.25f : -1e30f;
      float bm = fmaxf(x0, x1);
      bm = fmaxf(bm, __shfl_xor(bm, 1, 32));
      bm = fmaxf(bm, __shfl_xor(bm, 2, 32));
      bm = fmaxf(bm, __shfl_xor(bm, 4, 32));
      bm = fmaxf(bm, __shfl_xor(bm, 8, 32));
      float mn = fmaxf(mrow[j], bm);
      float sc = __expf(mrow[j] - mn);
      float p0 = __expf(x0 - mn);
      float p1 = __expf(x1 - mn);
      float ps = p0 + p1;
      ps += __shfl_xor(ps, 1, 32);
      ps += __shfl_xor(ps, 2, 32);
      ps += __shfl_xor(ps, 4, 32);
      ps += __shfl_xor(ps, 8, 32);
      srow[j] = srow[j] * sc + ps;
      mrow[j] = mn;
      acc.e[j] *= sc;
      Pt[(j + 8 * g) * 32 + m]      = (_Float16)p0;
      Pt[(j + 8 * g) * 32 + 16 + m] = (_Float16)p1;
    }

    asm volatile("s_wait_dscnt 0" ::: "memory");   // D-layout stores -> A-layout loads

    H16 pa;                                  // P as A-fragment (full K=32)
    pa.h[0] = *(const v8h*)(&Pt[m * 32 + 8 * g]);
    pa.h[1] = *(const v8h*)(&Pt[m * 32 + 16 + 8 * g]);

    // V as B-fragment from V^T: table order, 16B-aligned (kbase % 32 == 0)
    H16 va;
    va.h[0] = *(const v8h*)(Vts + vtb + (size_t)(kbase + 8 * g));
    va.h[1] = *(const v8h*)(Vts + vtb + (size_t)(kbase + 16 + 8 * g));

    acc.v = wmma_f16(pa.v, va.v, acc.v);
  }

  // singleton "diag" column per row: pick row q=1+p vs key 251+p (Q1),
  // delivery row q=251+p vs key 1+p (Q4).
#pragma unroll
  for (int j = 0; j < 8; ++j) {
    int grow = r0 + j + 8 * g;
    bool pick = (grow >= 1) && (grow <= 250);
    bool del  = (grow >= 251) && (grow <= 500);
    int dkey = pick ? (grow + 250) : (del ? (grow - 250) : 0);
    H16 dqv, krv;
    dqv.v = *(const v16h*)(Dgs + (hb512 + grow) * 16);
    krv.v = *(const v16h*)(Ks  + (hb512 + dkey) * 16);
    float s = 0.f;
#pragma unroll
    for (int kk = 0; kk < 16; ++kk) s += (float)dqv.e[kk] * (float)krv.e[kk];
    s *= 0.25f;
    float sd = (pick || del) ? s : -1e30f;
    float mn = fmaxf(mrow[j], sd);
    float sc = __expf(mrow[j] - mn);
    float pd = __expf(sd - mn);
    srow[j] = srow[j] * sc + pd;
    mrow[j] = mn;
    float vv = (float)Vts[vtb + dkey];
    acc.e[j] = acc.e[j] * sc + pd * vv;
  }

  // normalize and store heads as f16 [b][n][h*16] (contraction-contiguous)
#pragma unroll
  for (int j = 0; j < 8; ++j) {
    int grow = r0 + j + 8 * g;
    if (grow < GG) {
      float r = 1.0f / srow[j];
      Hd[(((size_t)b * RPAD + grow) * HH + h) * 16 + m] = (_Float16)(acc.e[j] * r);
    }
  }
}

// ---------------------------------------------------------------------------
// Kernel 3: out[b,n,e] = heads[b,n,hk] @ W_out[hk,e], contraction = 128.
// grid = B * 32 n-tiles * 8 e-tiles. B-fragments come from the pre-transposed
// f16 WoutT so every fragment load is a contiguous aligned vector load.
// ---------------------------------------------------------------------------
__global__ void __launch_bounds__(32) out_kernel(const _Float16* __restrict__ ws,
                                                 float* __restrict__ out) {
  const _Float16* Hd    = ws + 6 * SREG;
  const _Float16* WoutT = ws + 7 * SREG;
  const int l = threadIdx.x;
  const int m = l & 15;
  const int g = l >> 4;
  const int et = blockIdx.x & 7;
  const int r  = blockIdx.x >> 3;
  const int nt = r & 31;
  const int b  = r >> 5;
  const int n0 = nt * 16, e0 = et * 16;

  F8 acc;
#pragma unroll
  for (int i = 0; i < 8; ++i) acc.e[i] = 0.f;

#pragma unroll
  for (int c = 0; c < 4; ++c) {
    H16 a;
    const _Float16* ap = Hd + ((size_t)b * RPAD + (n0 + m)) * 128 + 32 * c + 8 * g;
    a.h[0] = *(const v8h*)ap;
    a.h[1] = *(const v8h*)(ap + 16);
    H16 bf;  // lane col n = e0+m; e<8 -> K = 32c + 8g + e, e>=8 -> K = 32c + 16 + 8g + e-8
    bf.h[0] = *(const v8h*)(WoutT + (size_t)(e0 + m) * 128 + 32 * c + 8 * g);
    bf.h[1] = *(const v8h*)(WoutT + (size_t)(e0 + m) * 128 + 32 * c + 16 + 8 * g);
    acc.v = wmma_f16(a.v, bf.v, acc.v);
  }

#pragma unroll
  for (int j = 0; j < 8; ++j) {
    int grow = n0 + j + 8 * g;
    if (grow < GG) out[((size_t)b * GG + grow) * EE + e0 + m] = acc.e[j];
  }
}

// ---------------------------------------------------------------------------
extern "C" void kernel_launch(void* const* d_in, const int* in_sizes, int n_in,
                              void* d_out, int out_size, void* d_ws, size_t ws_size,
                              hipStream_t stream) {
  const float* q = (const float*)d_in[0];
  WPtrs wp;
  for (int i = 0; i < 9; ++i) wp.w[i] = (const float*)d_in[1 + i];
  const float* Wout = (const float*)d_in[10];
  _Float16* ws = (_Float16*)d_ws;       // needs (7*SREG + 16384)*2 ~= 14.03 MB
  float* out = (float*)d_out;

  dim3 blk(32);
  prep_wout<<<16, blk, 0, stream>>>(Wout, ws);
  proj_kernel<<<HH * BB * 32, blk, 0, stream>>>(q, wp, ws);
  attn_kernel<<<HH * BB * 32, blk, 0, stream>>>(ws);
  out_kernel<<<BB * 32 * 8, blk, 0, stream>>>(ws, out);
}
